// SpikingTransformer_57423712748061
// MI455X (gfx1250) — compile-verified, matrix operand order checked by reading, not run on hardware
//
#include <hip/hip_runtime.h>
#include <hip/hip_bf16.h>
#include <stdint.h>

typedef _Float16 half_t;
typedef __attribute__((ext_vector_type(16))) _Float16 v16h;
typedef __attribute__((ext_vector_type(8)))  float    v8f;

constexpr int kT     = 4;
constexpr int kB     = 32;
constexpr int kTB    = 128;   // T*B
constexpr int kC     = 384;
constexpr int kHeads = 12;
constexpr int kDh    = 32;    // C / heads
constexpr int kN     = 196;   // 14*14
constexpr int kHd    = 1536;  // 4*C

union FragA { v16h h; unsigned u[8]; };

// K index (pair base) for half-pair v (0..7) of a 16x32 f16 WMMA fragment,
// per CDNA5 ISA 7.12.2: lanes 0-15 hold K={0..7,16..23}, lanes 16-31 hold +8.
__device__ __forceinline__ int kbase(int v, int lane) {
  return ((v < 4) ? 2 * v : 2 * v + 8) + ((lane >= 16) ? 8 : 0);
}

// Low 32 bits of a generic pointer to __shared__ == LDS byte offset
// (flat LDS aperture).
__device__ __forceinline__ unsigned lds_off(const void* p) {
  return (unsigned)(uintptr_t)p;
}

// ---------------------------------------------------------------- convert
__global__ void k_cvt_f16(const float* __restrict__ in, half_t* __restrict__ out, int n) {
  int i = blockIdx.x * blockDim.x + threadIdx.x;
  if (i < n) out[i] = (half_t)in[i];
}

// ---------------------------------------------------------------- GEMM
// Out[b][m][n] = sum_k A[m][k] * B[b][k][n] (+ bias[m])
// A: MxK f16 row-major (weights).  B: kTB batches of Kx196 f16.  Out f32.
// Block tile: 128(M) x 64(N), BK=32, 256 threads = 8 waves, each wave owns a
// 16-row strip and 4 N-subtiles -> 4 v_wmma_f32_16x16x32_f16 per k-step.
// A tile staged memory->LDS with global_load_async_to_lds_b128 (ASYNCcnt);
// B tile staged transposed via regular loads + ds_store so fragment pair
// reads are contiguous; strides 40 halves (80B) keep b128 LDS accesses
// 16B-aligned so the compiler can merge fragment reads into ds_load_b128.
__global__ __launch_bounds__(256)
void k_gemm(const half_t* __restrict__ A, const half_t* __restrict__ B,
            const float* __restrict__ bias, float* __restrict__ Out,
            int M, int K) {
  constexpr int SA = 40;
  constexpr int SB = 40;
  __shared__ __align__(16) half_t As[128 * SA];
  __shared__ __align__(16) half_t Bs[64 * SB];
  const int m0 = blockIdx.x * 128;
  const int n0 = blockIdx.y * 64;
  const int bb = blockIdx.z;
  const int tid  = threadIdx.x;
  const int wave = tid >> 5, lane = tid & 31, lm = lane & 15;
  const half_t* Bbase = B + (size_t)bb * K * kN;

  v8f acc[4] = {};
  const int ksteps = K / 32;
  for (int kt = 0; kt < ksteps; ++kt) {
    const int kk = kt * 32;
    __syncthreads();
    if (kt + 1 < ksteps && lane == 0)
      __builtin_prefetch(Bbase + (size_t)(kk + 32) * kN + n0, 0, 1);
    // stage A tile 128x32: async DMA directly to LDS (two b128 per thread)
#pragma unroll
    for (int it = 0; it < 2; ++it) {
      int chunk = tid + it * 256;
      int row = chunk >> 2, c8 = (chunk & 3) * 8;
      unsigned dst = lds_off(&As[row * SA + c8]);
      unsigned long long src =
          (unsigned long long)(uintptr_t)(A + (size_t)(m0 + row) * K + kk + c8);
      asm volatile("global_load_async_to_lds_b128 %0, %1, off"
                   :: "v"(dst), "v"(src) : "memory");
    }
    // stage B tile 32x64, transposed into Bs[n][k] (pairs along k contiguous)
#pragma unroll
    for (int it = 0; it < 2; ++it) {
      int chunk = tid + it * 256;
      int kr = chunk >> 4, nc = (chunk & 15) * 4;
      int n = n0 + nc;
      half_t hv[4];
      if (n + 3 < kN) {
        union { uint2 u; half_t h[4]; } tmp;
        tmp.u = *reinterpret_cast<const uint2*>(Bbase + (size_t)(kk + kr) * kN + n);
        hv[0] = tmp.h[0]; hv[1] = tmp.h[1]; hv[2] = tmp.h[2]; hv[3] = tmp.h[3];
      } else {
#pragma unroll
        for (int i = 0; i < 4; ++i)
          hv[i] = (n + i < kN) ? Bbase[(size_t)(kk + kr) * kN + n + i] : (half_t)0.0f;
      }
#pragma unroll
      for (int i = 0; i < 4; ++i) Bs[(nc + i) * SB + kr] = hv[i];
    }
    asm volatile("s_wait_asynccnt 0x0" ::: "memory");  // A tile landed in LDS
    __syncthreads();

    FragA a;
#pragma unroll
    for (int v = 0; v < 8; ++v)
      a.u[v] = *reinterpret_cast<const unsigned*>(&As[(wave * 16 + lm) * SA + kbase(v, lane)]);
#pragma unroll
    for (int s = 0; s < 4; ++s) {
      FragA b;
#pragma unroll
      for (int v = 0; v < 8; ++v)
        b.u[v] = *reinterpret_cast<const unsigned*>(&Bs[(s * 16 + lm) * SB + kbase(v, lane)]);
      acc[s] = __builtin_amdgcn_wmma_f32_16x16x32_f16(false, a.h, false, b.h,
                                                      (short)0, acc[s], false, false);
    }
  }

  const int mrow = m0 + wave * 16 + ((lane >= 16) ? 8 : 0);
#pragma unroll
  for (int s = 0; s < 4; ++s) {
    int n = n0 + s * 16 + lm;
    if (n < kN) {
#pragma unroll
      for (int r = 0; r < 8; ++r) {
        int m = mrow + r;
        float bv = bias ? bias[m] : 0.0f;
        Out[((size_t)bb * M + m) * kN + n] = acc[s][r] + bv;
      }
    }
  }
}

// ---------------------------------------------------------------- BN stats
// One block per channel: mean/var over (TB, N); emit scale=gamma*rstd,
// shift=beta-mean*scale so LIF kernels apply a single FMA.
__global__ __launch_bounds__(256)
void k_bn_stats(const float* __restrict__ y, const float* __restrict__ gamma,
                const float* __restrict__ beta, float* __restrict__ scale,
                float* __restrict__ shift, int M) {
  __shared__ float rs[256], rq[256];
  const int c = blockIdx.x, tid = threadIdx.x;
  const int cnt = kTB * kN;
  float s = 0.f, q = 0.f;
  for (int i = tid; i < cnt; i += 256) {
    int b = i / kN, n = i - b * kN;
    float v = y[((size_t)b * M + c) * kN + n];
    s += v; q += v * v;
  }
  rs[tid] = s; rq[tid] = q;
  __syncthreads();
  for (int off = 128; off > 0; off >>= 1) {
    if (tid < off) { rs[tid] += rs[tid + off]; rq[tid] += rq[tid + off]; }
    __syncthreads();
  }
  if (tid == 0) {
    float mean = rs[0] / cnt;
    float var  = rq[0] / cnt - mean * mean;
    float rstd = rsqrtf(var + 1e-5f);
    float sc = gamma[c] * rstd;
    scale[c] = sc;
    shift[c] = beta[c] - mean * sc;
  }
}

// ---------------------------------------------------------------- LIF
// layout 0: natural (tb, c, n)
// layout 1: head-major  tb*C*N + h*N*32 + n*32 + d          (A-frag friendly)
// layout 2: pair-interleaved tb*C*N + h*N*32 + (n/2)*64 + d*2 + (n&1)  (B-frag friendly)
__global__ void k_lif_spike(const float* __restrict__ y, const float* __restrict__ scale,
                            const float* __restrict__ shift, float thr,
                            half_t* __restrict__ out, int M, int layout) {
  int idx = blockIdx.x * blockDim.x + threadIdx.x;
  int total = kB * M * kN;
  if (idx >= total) return;
  int n = idx % kN;
  int t2 = idx / kN;
  int c = t2 % M;
  int b = t2 / M;
  float sc = scale ? scale[c] : 1.0f;
  float sh = shift ? shift[c] : 0.0f;
  float mem = 0.f;
#pragma unroll
  for (int t = 0; t < kT; ++t) {
    size_t tb = (size_t)(t * kB + b);
    float v = y[(tb * M + c) * kN + n] * sc + sh;
    mem = 0.5f * (mem + v);                 // mem += (x - mem)/TAU, TAU=2
    float sp = (mem >= thr) ? 1.0f : 0.0f;
    size_t o;
    if (layout == 0)      o = (tb * M + c) * kN + n;
    else if (layout == 1) o = tb * (size_t)(kC * kN) + (size_t)(c >> 5) * (kN * 32)
                              + (size_t)n * 32 + (c & 31);
    else                  o = tb * (size_t)(kC * kN) + (size_t)(c >> 5) * (kN * 32)
                              + (size_t)(n >> 1) * 64 + (c & 31) * 2 + (n & 1);
    out[o] = (half_t)sp;
    mem *= (1.0f - sp);                     // hard reset
  }
}

// LIF (thr=1) + residual add: outf = xres + spike; optional f16 copy for next GEMM.
__global__ void k_lif_res(const float* __restrict__ y, const float* __restrict__ scale,
                          const float* __restrict__ shift, const float* __restrict__ xres,
                          float* __restrict__ outf, half_t* __restrict__ outh, int M) {
  int idx = blockIdx.x * blockDim.x + threadIdx.x;
  int total = kB * M * kN;
  if (idx >= total) return;
  int n = idx % kN;
  int t2 = idx / kN;
  int c = t2 % M;
  int b = t2 / M;
  float sc = scale[c], sh = shift[c];
  float mem = 0.f;
#pragma unroll
  for (int t = 0; t < kT; ++t) {
    size_t g = ((size_t)(t * kB + b) * M + c) * kN + n;
    float v = y[g] * sc + sh;
    mem = 0.5f * (mem + v);
    float sp = (mem >= 1.0f) ? 1.0f : 0.0f;
    float r = xres[g] + sp;
    outf[g] = r;
    if (outh) outh[g] = (half_t)r;
    mem *= (1.0f - sp);
  }
}

// ---------------------------------------------------------------- attention
// One wave per (tb, head): kv = k^T v (32x32, K=196 padded to 224) then
// attn = q * kv * 0.125, written transposed as (tb, c=h*32+e, n).
__global__ __launch_bounds__(32)
void k_attn(const half_t* __restrict__ sq, const half_t* __restrict__ sk,
            const half_t* __restrict__ sv, float* __restrict__ attnf) {
  __shared__ half_t kvs[16 * 64];  // kv staged as B-fragment layout: (d/2)*64 + e*2 + (d&1)
  const int bh = blockIdx.x;
  const int tb = bh / kHeads, h = bh % kHeads;
  const int lane = threadIdx.x & 31, lm = lane & 15;
  const int hi8 = (lane >= 16) ? 8 : 0;
  const half_t* skb = sk + ((size_t)tb * kC + h * kDh) * kN;               // (d, n) rows
  const half_t* svb = sv + (size_t)tb * kC * kN + (size_t)h * kN * kDh;    // pair-interleaved
  const half_t* sqb = sq + (size_t)tb * kC * kN + (size_t)h * kN * kDh;    // (n, d) rows

  v8f acc[2][2] = {};
  for (int ks = 0; ks < 7; ++ks) {
    int kb = ks * 32;
    FragA a[2], b[2];
#pragma unroll
    for (int dt = 0; dt < 2; ++dt)
#pragma unroll
      for (int v = 0; v < 8; ++v) {
        int K = kb + kbase(v, lane);
        a[dt].u[v] = (K < kN)
          ? *reinterpret_cast<const unsigned*>(&skb[(size_t)(dt * 16 + lm) * kN + K]) : 0u;
      }
#pragma unroll
    for (int et = 0; et < 2; ++et)
#pragma unroll
      for (int v = 0; v < 8; ++v) {
        int K = kb + kbase(v, lane);
        b[et].u[v] = (K < kN)
          ? *reinterpret_cast<const unsigned*>(&svb[(size_t)(K >> 1) * 64 + (et * 16 + lm) * 2]) : 0u;
      }
#pragma unroll
    for (int dt = 0; dt < 2; ++dt)
#pragma unroll
      for (int et = 0; et < 2; ++et)
        acc[dt][et] = __builtin_amdgcn_wmma_f32_16x16x32_f16(false, a[dt].h, false, b[et].h,
                                                             (short)0, acc[dt][et], false, false);
  }
#pragma unroll
  for (int dt = 0; dt < 2; ++dt)
#pragma unroll
    for (int et = 0; et < 2; ++et)
#pragma unroll
      for (int r = 0; r < 8; ++r) {
        int d = dt * 16 + r + hi8;
        int e = et * 16 + lm;
        kvs[(d >> 1) * 64 + e * 2 + (d & 1)] = (half_t)acc[dt][et][r];
      }
  __syncthreads();

  float* ao = attnf + ((size_t)tb * kC + h * kDh) * kN;
  for (int mt = 0; mt < 13; ++mt) {
    int n0r = mt * 16;
    int row = n0r + lm;
    FragA a;
#pragma unroll
    for (int v = 0; v < 8; ++v)
      a.u[v] = (row < kN)
        ? *reinterpret_cast<const unsigned*>(&sqb[(size_t)row * kDh + kbase(v, lane)]) : 0u;
#pragma unroll
    for (int et = 0; et < 2; ++et) {
      FragA b;
#pragma unroll
      for (int v = 0; v < 8; ++v) {
        int K = kbase(v, lane);
        b.u[v] = *reinterpret_cast<const unsigned*>(&kvs[(K >> 1) * 64 + (et * 16 + lm) * 2]);
      }
      v8f c = {};
      c = __builtin_amdgcn_wmma_f32_16x16x32_f16(false, a.h, false, b.h, (short)0, c, false, false);
#pragma unroll
      for (int r = 0; r < 8; ++r) {
        int nr = n0r + r + hi8;
        if (nr < kN) {
          int e = et * 16 + lm;
          ao[(size_t)e * kN + nr] = 0.125f * c[r];   // SCALE
        }
      }
    }
  }
}

// ---------------------------------------------------------------- host
extern "C" void kernel_launch(void* const* d_in, const int* in_sizes, int n_in,
                              void* d_out, int out_size, void* d_ws, size_t ws_size,
                              hipStream_t stream) {
  (void)in_sizes; (void)n_in; (void)out_size; (void)ws_size;
  const float* x          = (const float*)d_in[0];
  const float* q_w        = (const float*)d_in[1];
  const float* q_gamma    = (const float*)d_in[2];
  const float* q_beta     = (const float*)d_in[3];
  const float* k_w        = (const float*)d_in[4];
  const float* k_gamma    = (const float*)d_in[5];
  const float* k_beta     = (const float*)d_in[6];
  const float* v_w        = (const float*)d_in[7];
  const float* v_gamma    = (const float*)d_in[8];
  const float* v_beta     = (const float*)d_in[9];
  const float* proj_w     = (const float*)d_in[10];
  const float* proj_b     = (const float*)d_in[11];
  const float* proj_gamma = (const float*)d_in[12];
  const float* proj_beta  = (const float*)d_in[13];
  const float* fc1_w      = (const float*)d_in[14];
  const float* fc1_b      = (const float*)d_in[15];
  const float* bn1_gamma  = (const float*)d_in[16];
  const float* bn1_beta   = (const float*)d_in[17];
  const float* fc2_w      = (const float*)d_in[18];
  const float* fc2_b      = (const float*)d_in[19];
  const float* bn2_gamma  = (const float*)d_in[20];
  const float* bn2_beta   = (const float*)d_in[21];

  char* ws = (char*)d_ws;
  const size_t P = (size_t)kTB * kC * kN;    // 9,633,792 elements
  const size_t Q = (size_t)kTB * kHd * kN;   // 38,535,168 elements
  size_t off = 0;
  auto alloc = [&](size_t bytes) { size_t o = off; off += (bytes + 255) & ~(size_t)255; return o; };

  size_t o_wq = alloc((size_t)kC * kC * 2);
  size_t o_wk = alloc((size_t)kC * kC * 2);
  size_t o_wv = alloc((size_t)kC * kC * 2);
  size_t o_wp = alloc((size_t)kC * kC * 2);
  size_t o_w1 = alloc((size_t)kHd * kC * 2);
  size_t o_w2 = alloc((size_t)kC * kHd * 2);
  size_t o_scq = alloc(kC * 4),  o_shq = alloc(kC * 4);
  size_t o_sck = alloc(kC * 4),  o_shk = alloc(kC * 4);
  size_t o_scv = alloc(kC * 4),  o_shv = alloc(kC * 4);
  size_t o_scp = alloc(kC * 4),  o_shp = alloc(kC * 4);
  size_t o_sc1 = alloc(kHd * 4), o_sh1 = alloc(kHd * 4);
  size_t o_sc2 = alloc(kC * 4),  o_sh2 = alloc(kC * 4);
  // aliased regions (liveness-based reuse)
  size_t o_big = alloc(4 * Q);   // yq|yk|yv early; yfc1 later
  size_t o_xh  = alloc(2 * P);   // xh; s_attn later
  size_t o_s   = alloc(6 * P);   // sq|sk|sv; x1h later
  size_t o_at  = alloc(4 * P);   // attnf; yproj; yfc2
  size_t o_x1  = alloc(4 * P);   // x1 (residual), live to end
  size_t o_sf  = alloc(2 * Q);   // fc1 spikes

  half_t* wqh  = (half_t*)(ws + o_wq);
  half_t* wkh  = (half_t*)(ws + o_wk);
  half_t* wvh  = (half_t*)(ws + o_wv);
  half_t* wph  = (half_t*)(ws + o_wp);
  half_t* w1h  = (half_t*)(ws + o_w1);
  half_t* w2h  = (half_t*)(ws + o_w2);
  float*  yq   = (float*)(ws + o_big);
  float*  yk   = yq + P;
  float*  yv   = yk + P;
  float*  yfc1 = (float*)(ws + o_big);
  half_t* xh     = (half_t*)(ws + o_xh);
  half_t* s_attn = (half_t*)(ws + o_xh);
  half_t* sq   = (half_t*)(ws + o_s);
  half_t* sk   = sq + P;
  half_t* sv   = sk + P;
  half_t* x1h  = (half_t*)(ws + o_s);
  float*  attnf = (float*)(ws + o_at);
  float*  yproj = (float*)(ws + o_at);
  float*  yfc2  = (float*)(ws + o_at);
  float*  x1f   = (float*)(ws + o_x1);
  half_t* sfc1  = (half_t*)(ws + o_sf);

  auto grid1 = [](size_t n) { return dim3((unsigned)((n + 255) / 256)); };

  // stage 0: f32 -> f16 conversions
  k_cvt_f16<<<grid1(P), 256, 0, stream>>>(x, xh, (int)P);
  k_cvt_f16<<<grid1(kC * kC), 256, 0, stream>>>(q_w, wqh, kC * kC);
  k_cvt_f16<<<grid1(kC * kC), 256, 0, stream>>>(k_w, wkh, kC * kC);
  k_cvt_f16<<<grid1(kC * kC), 256, 0, stream>>>(v_w, wvh, kC * kC);
  k_cvt_f16<<<grid1(kC * kC), 256, 0, stream>>>(proj_w, wph, kC * kC);
  k_cvt_f16<<<grid1((size_t)kHd * kC), 256, 0, stream>>>(fc1_w, w1h, kHd * kC);
  k_cvt_f16<<<grid1((size_t)kC * kHd), 256, 0, stream>>>(fc2_w, w2h, kC * kHd);

  const dim3 gG(kC / 128, 4, kTB);    // (3,4,128)
  const dim3 g1(kHd / 128, 4, kTB);   // (12,4,128)
  const int nl  = kB * kC * kN;
  const int nl2 = kB * kHd * kN;

  // q/k/v GEMMs, BN stats, LIF -> spikes in attention-friendly layouts
  k_gemm<<<gG, 256, 0, stream>>>(wqh, xh, nullptr, yq, kC, kC);
  k_gemm<<<gG, 256, 0, stream>>>(wkh, xh, nullptr, yk, kC, kC);
  k_gemm<<<gG, 256, 0, stream>>>(wvh, xh, nullptr, yv, kC, kC);
  k_bn_stats<<<kC, 256, 0, stream>>>(yq, q_gamma, q_beta, (float*)(ws + o_scq), (float*)(ws + o_shq), kC);
  k_bn_stats<<<kC, 256, 0, stream>>>(yk, k_gamma, k_beta, (float*)(ws + o_sck), (float*)(ws + o_shk), kC);
  k_bn_stats<<<kC, 256, 0, stream>>>(yv, v_gamma, v_beta, (float*)(ws + o_scv), (float*)(ws + o_shv), kC);
  k_lif_spike<<<grid1(nl), 256, 0, stream>>>(yq, (float*)(ws + o_scq), (float*)(ws + o_shq), 1.0f, sq, kC, 1);
  k_lif_spike<<<grid1(nl), 256, 0, stream>>>(yk, (float*)(ws + o_sck), (float*)(ws + o_shk), 1.0f, sk, kC, 0);
  k_lif_spike<<<grid1(nl), 256, 0, stream>>>(yv, (float*)(ws + o_scv), (float*)(ws + o_shv), 1.0f, sv, kC, 2);

  // attention + LIF(0.5)
  k_attn<<<kTB * kHeads, 32, 0, stream>>>(sq, sk, sv, attnf);
  k_lif_spike<<<grid1(nl), 256, 0, stream>>>(attnf, nullptr, nullptr, 0.5f, s_attn, kC, 0);

  // proj + BN + LIF + residual (x1 = x + spikes)
  k_gemm<<<gG, 256, 0, stream>>>(wph, s_attn, proj_b, yproj, kC, kC);
  k_bn_stats<<<kC, 256, 0, stream>>>(yproj, proj_gamma, proj_beta, (float*)(ws + o_scp), (float*)(ws + o_shp), kC);
  k_lif_res<<<grid1(nl), 256, 0, stream>>>(yproj, (float*)(ws + o_scp), (float*)(ws + o_shp), x, x1f, x1h, kC);

  // MLP
  k_gemm<<<g1, 256, 0, stream>>>(w1h, x1h, fc1_b, yfc1, kHd, kC);
  k_bn_stats<<<kHd, 256, 0, stream>>>(yfc1, bn1_gamma, bn1_beta, (float*)(ws + o_sc1), (float*)(ws + o_sh1), kHd);
  k_lif_spike<<<grid1(nl2), 256, 0, stream>>>(yfc1, (float*)(ws + o_sc1), (float*)(ws + o_sh1), 1.0f, sfc1, kHd, 0);
  k_gemm<<<gG, 256, 0, stream>>>(w2h, sfc1, fc2_b, yfc2, kC, kHd);
  k_bn_stats<<<kC, 256, 0, stream>>>(yfc2, bn2_gamma, bn2_beta, (float*)(ws + o_sc2), (float*)(ws + o_sh2), kC);
  k_lif_res<<<grid1(nl), 256, 0, stream>>>(yfc2, (float*)(ws + o_sc2), (float*)(ws + o_sh2), x1f,
                                           (float*)d_out, nullptr, kC);
}
